// DVAE_50268297232752
// MI455X (gfx1250) — matrix operand order repaired and checked by reading, not compiled, not run on hardware
//
#include <hip/hip_runtime.h>
#include <hip/hip_bf16.h>

// ---------------------------------------------------------------------------
// Problem constants (B, N, H, L, T) = (64, 1024, 512, 256, 32)
// ---------------------------------------------------------------------------
#define BB 64
#define NN 1024
#define HH 512
#define LL 256
#define TT 32

typedef __attribute__((ext_vector_type(16))) __bf16 v16bf;
typedef __attribute__((ext_vector_type(8)))  __bf16 v8bf;
typedef __attribute__((ext_vector_type(8)))  float  v8f;

// ---------------------------------------------------------------------------
// K0: convert W1/W2 (fp32, [K][N] row-major) -> bf16 transposed [N][K]
// so B-fragments become contiguous 16-element runs per lane.
// ---------------------------------------------------------------------------
__global__ void convert_weights_kernel(const float* __restrict__ w1,
                                       const float* __restrict__ w2,
                                       __bf16* __restrict__ w1t,
                                       __bf16* __restrict__ w2t) {
    int id = blockIdx.x * blockDim.x + threadIdx.x;   // over 512*512
    if (id >= HH * HH) return;
    int k = id >> 9;          // row (K) in source
    int n = id & (HH - 1);    // col (N) in source
    w1t[n * HH + k] = (__bf16)w1[id];
    w2t[n * HH + k] = (__bf16)w2[id];
}

__global__ void zero_kernel(float* __restrict__ p, int n) {
    int i = blockIdx.x * blockDim.x + threadIdx.x;
    if (i < n) p[i] = 0.0f;
}

// ---------------------------------------------------------------------------
// K1: fused GNN (Linear->ReLU->Linear) + mean pool, WMMA bf16, f32 accum.
// One workgroup = 32 node rows (all in one batch), 512 threads = 16 waves.
// Wave w: m-tile = w>>3 (16 rows), n-slab = (w&7)*64 (4 WMMA n-tiles).
// ---------------------------------------------------------------------------
__global__ __launch_bounds__(512) void gnn_pool_kernel(
    const float*  __restrict__ X,        // [B*N, H] fp32
    const __bf16* __restrict__ w1t,      // [N][K] bf16
    const float*  __restrict__ b1,
    const __bf16* __restrict__ w2t,      // [N][K] bf16
    const float*  __restrict__ b2,
    float*        __restrict__ h_pooled) // [B, H], pre-zeroed
{
    __shared__ __align__(16) __bf16 sX[32][HH];
    __shared__ __align__(16) __bf16 sY[32][HH];

    const int tid      = threadIdx.x;
    const int wave     = tid >> 5;
    const int lane     = tid & 31;
    const int row_base = blockIdx.x * 32;
    const int batch    = row_base >> 10;        // N = 1024 rows per batch

    // ---- stage X tile into LDS as bf16 (coalesced float4 loads) ----
    {
        const int r  = tid >> 4;                // 0..31
        const int c0 = (tid & 15) * 32;         // 32 floats per thread
        const float4* gp =
            reinterpret_cast<const float4*>(X + (size_t)(row_base + r) * HH + c0);
#pragma unroll
        for (int i = 0; i < 8; ++i) {
            float4 v = gp[i];
            sX[r][c0 + i * 4 + 0] = (__bf16)v.x;
            sX[r][c0 + i * 4 + 1] = (__bf16)v.y;
            sX[r][c0 + i * 4 + 2] = (__bf16)v.z;
            sX[r][c0 + i * 4 + 3] = (__bf16)v.w;
        }
    }
    __syncthreads();

    const int mt    = wave >> 3;        // 0..1  : m-tile
    const int nb    = (wave & 7) * 64;  // base output column of this wave
    const int lmod  = lane & 15;
    const int lhalf = lane >> 4;        // 0/1
    const int a_off = lhalf * 16;       // A-frag K-chunk select (contiguous, matches B)
    const int b_off = lhalf * 16;       // B-frag K-chunk select
    const int mrow  = mt * 16 + lmod;   // A row this lane reads

    v8f acc[4];

    // ================= layer 1: Y = relu(X*W1 + b1) =================
#pragma unroll
    for (int t = 0; t < 4; ++t) {
        float bv = b1[nb + t * 16 + lmod];
#pragma unroll
        for (int r = 0; r < 8; ++r) acc[t][r] = bv;
    }

#pragma unroll
    for (int kb = 0; kb < HH; kb += 32) {
        v8bf alo = *reinterpret_cast<const v8bf*>(&sX[mrow][kb + a_off]);
        v8bf ahi = *reinterpret_cast<const v8bf*>(&sX[mrow][kb + a_off + 8]);
        v16bf a = __builtin_shufflevector(alo, ahi,
                      0,1,2,3,4,5,6,7,8,9,10,11,12,13,14,15);
#pragma unroll
        for (int t = 0; t < 4; ++t) {
            const __bf16* bp = w1t + (size_t)(nb + t * 16 + lmod) * HH + kb + b_off;
            v8bf blo = *reinterpret_cast<const v8bf*>(bp);
            v8bf bhi = *reinterpret_cast<const v8bf*>(bp + 8);
            v16bf b = __builtin_shufflevector(blo, bhi,
                          0,1,2,3,4,5,6,7,8,9,10,11,12,13,14,15);
            acc[t] = __builtin_amdgcn_wmma_f32_16x16x32_bf16(
                false, a, false, b, (short)0, acc[t], false, false);
        }
    }

    // ReLU + spill to LDS as bf16 (C layout: M = r + 8*lhalf, N = nb+16t+lmod)
#pragma unroll
    for (int t = 0; t < 4; ++t)
#pragma unroll
        for (int r = 0; r < 8; ++r) {
            float v = acc[t][r];
            v = v > 0.0f ? v : 0.0f;
            sY[mt * 16 + lhalf * 8 + r][nb + t * 16 + lmod] = (__bf16)v;
        }
    __syncthreads();

    // ================= layer 2: H = Y*W2 + b2, then mean-pool =================
#pragma unroll
    for (int t = 0; t < 4; ++t) {
        float bv = b2[nb + t * 16 + lmod];
#pragma unroll
        for (int r = 0; r < 8; ++r) acc[t][r] = bv;
    }

#pragma unroll
    for (int kb = 0; kb < HH; kb += 32) {
        v8bf alo = *reinterpret_cast<const v8bf*>(&sY[mrow][kb + a_off]);
        v8bf ahi = *reinterpret_cast<const v8bf*>(&sY[mrow][kb + a_off + 8]);
        v16bf a = __builtin_shufflevector(alo, ahi,
                      0,1,2,3,4,5,6,7,8,9,10,11,12,13,14,15);
#pragma unroll
        for (int t = 0; t < 4; ++t) {
            const __bf16* bp = w2t + (size_t)(nb + t * 16 + lmod) * HH + kb + b_off;
            v8bf blo = *reinterpret_cast<const v8bf*>(bp);
            v8bf bhi = *reinterpret_cast<const v8bf*>(bp + 8);
            v16bf b = __builtin_shufflevector(blo, bhi,
                          0,1,2,3,4,5,6,7,8,9,10,11,12,13,14,15);
            acc[t] = __builtin_amdgcn_wmma_f32_16x16x32_bf16(
                false, a, false, b, (short)0, acc[t], false, false);
        }
    }

    // mean-pool partial: each lane-half holds 8 M-rows for a fixed column
#pragma unroll
    for (int t = 0; t < 4; ++t) {
        float s = 0.0f;
#pragma unroll
        for (int r = 0; r < 8; ++r) s += acc[t][r];
        atomicAdd(&h_pooled[batch * HH + nb + t * 16 + lmod], s * (1.0f / (float)NN));
    }
}

// ---------------------------------------------------------------------------
// K2: mu / logvar / reparameterized z.  One thread per (b, j), j < L.
// ---------------------------------------------------------------------------
__global__ void head_kernel(const float* __restrict__ hp,
                            const float* __restrict__ mu_w, const float* __restrict__ mu_b,
                            const float* __restrict__ lv_w, const float* __restrict__ lv_b,
                            const float* __restrict__ eps,
                            float* __restrict__ mu_out, float* __restrict__ lv_out,
                            float* __restrict__ z) {
    int id = blockIdx.x * blockDim.x + threadIdx.x;       // B*L
    if (id >= BB * LL) return;
    int b = id >> 8, j = id & (LL - 1);
    const float* h = hp + b * HH;
    float am = mu_b[j], al = lv_b[j];
    for (int k = 0; k < HH; ++k) {
        float hv = h[k];
        am += hv * mu_w[k * LL + j];
        al += hv * lv_w[k * LL + j];
    }
    mu_out[id] = am;
    lv_out[id] = al;
    z[id] = am + eps[id] * __expf(0.5f * al);
}

// ---------------------------------------------------------------------------
// K3: hd = relu(z * z_w + z_b).  One thread per (b, n), n < H.
// ---------------------------------------------------------------------------
__global__ void dec_z_kernel(const float* __restrict__ z,
                             const float* __restrict__ z_w, const float* __restrict__ z_b,
                             float* __restrict__ hd) {
    int id = blockIdx.x * blockDim.x + threadIdx.x;       // B*H
    if (id >= BB * HH) return;
    int b = id >> 9, n = id & (HH - 1);
    const float* zz = z + b * LL;
    float a = z_b[n];
    for (int k = 0; k < LL; ++k) a += zz[k] * z_w[k * HH + n];
    hd[id] = a > 0.0f ? a : 0.0f;
}

// ---------------------------------------------------------------------------
// K4: per-batch edge scalar e[b] = sigmoid(relu([hd,hd]*E1+b)*E2+b2) and
// node vector node_vec[b][t] = hd*nd_w + nd_b.  One block per batch.
// ---------------------------------------------------------------------------
__global__ __launch_bounds__(256) void edge_node_kernel(
    const float* __restrict__ hd,
    const float* __restrict__ nd_w, const float* __restrict__ nd_b,
    const float* __restrict__ e1_w, const float* __restrict__ e1_b,
    const float* __restrict__ e2_w, const float* __restrict__ e2_b,
    float* __restrict__ node_vec, float* __restrict__ e_out) {
    __shared__ float red[256];
    const int b = blockIdx.x, tid = threadIdx.x;
    const float* h = hd + b * HH;

    float partial = 0.0f;
#pragma unroll
    for (int rep = 0; rep < 2; ++rep) {
        int n = tid + rep * 256;
        float u = e1_b[n];
        for (int k = 0; k < HH; ++k)
            u += h[k] * (e1_w[k * HH + n] + e1_w[(k + HH) * HH + n]); // [hd,hd] concat
        if (u > 0.0f) partial += u * e2_w[n];
    }
    red[tid] = partial;
    __syncthreads();
    for (int s = 128; s > 0; s >>= 1) {
        if (tid < s) red[tid] += red[tid + s];
        __syncthreads();
    }
    if (tid == 0) e_out[b] = 1.0f / (1.0f + __expf(-(red[0] + e2_b[0])));

    if (tid < TT) {
        float a = nd_b[tid];
        for (int k = 0; k < HH; ++k) a += h[k] * nd_w[k * TT + tid];
        node_vec[b * TT + tid] = a;
    }
}

// ---------------------------------------------------------------------------
// K5a: node_logits[b][i][t] = node_vec[b][t]   (broadcast over i)
// ---------------------------------------------------------------------------
__global__ void bcast_nodes_kernel(const float* __restrict__ nv,
                                   float* __restrict__ out) {
    long idx = (long)blockIdx.x * blockDim.x + threadIdx.x;   // B*N*T
    if (idx >= (long)BB * NN * TT) return;
    int t = (int)(idx & (TT - 1));
    int b = (int)(idx >> 15);          // N*T = 32768 per batch
    out[idx] = nv[b * TT + t];
}

// ---------------------------------------------------------------------------
// K5b: edge_probs[b][i][j] = e[b] * (i > j)  -- strict lower triangle,
// float4-vectorized streaming stores (268 MB, pure bandwidth).
// ---------------------------------------------------------------------------
__global__ void bcast_edges_kernel(const float* __restrict__ e,
                                   float4* __restrict__ out) {
    long g = (long)blockIdx.x * blockDim.x + threadIdx.x;     // B*N*(N/4)
    if (g >= (long)BB * NN * (NN / 4)) return;
    int j4 = (int)(g & 255) * 4;
    int i  = (int)((g >> 8) & (NN - 1));
    int b  = (int)(g >> 18);
    float ev = e[b];
    float4 v;
    v.x = (i > j4 + 0) ? ev : 0.0f;
    v.y = (i > j4 + 1) ? ev : 0.0f;
    v.z = (i > j4 + 2) ? ev : 0.0f;
    v.w = (i > j4 + 3) ? ev : 0.0f;
    out[g] = v;
}

// ---------------------------------------------------------------------------
// Launcher
// ---------------------------------------------------------------------------
extern "C" void kernel_launch(void* const* d_in, const int* in_sizes, int n_in,
                              void* d_out, int out_size, void* d_ws, size_t ws_size,
                              hipStream_t stream) {
    // inputs in setup_inputs() order
    const float* X     = (const float*)d_in[0];   // node_features [B,N,H]
    // d_in[1] = adj_matrix (unused)
    const float* eps   = (const float*)d_in[2];   // [B,L]
    const float* gw1   = (const float*)d_in[3];
    const float* gb1   = (const float*)d_in[4];
    const float* gw2   = (const float*)d_in[5];
    const float* gb2   = (const float*)d_in[6];
    const float* mu_w  = (const float*)d_in[7];
    const float* mu_b  = (const float*)d_in[8];
    const float* lv_w  = (const float*)d_in[9];
    const float* lv_b  = (const float*)d_in[10];
    const float* z_w   = (const float*)d_in[11];
    const float* z_b   = (const float*)d_in[12];
    const float* nd_w  = (const float*)d_in[13];
    const float* nd_b  = (const float*)d_in[14];
    const float* e1_w  = (const float*)d_in[15];
    const float* e1_b  = (const float*)d_in[16];
    const float* e2_w  = (const float*)d_in[17];
    const float* e2_b  = (const float*)d_in[18];

    // workspace layout (bytes)
    char* ws = (char*)d_ws;
    __bf16* w1t     = (__bf16*)(ws + 0);                    // 512*512*2
    __bf16* w2t     = (__bf16*)(ws + 524288);               // 512*512*2
    float*  h_pool  = (float*) (ws + 1048576);              // 64*512*4
    float*  z       = (float*) (ws + 1179648);              // 64*256*4
    float*  hd      = (float*) (ws + 1245184);              // 64*512*4
    float*  nodevec = (float*) (ws + 1376256);              // 64*32*4
    float*  e_scal  = (float*) (ws + 1384448);              // 64*4

    // output layout (floats, tuple concat order)
    float* out         = (float*)d_out;
    float* node_logits = out;                               // B*N*T
    float* edge_probs  = out + (long)BB * NN * TT;          // B*N*N
    float* mu_out      = out + (long)BB * NN * TT + (long)BB * NN * NN;
    float* lv_out      = mu_out + (long)BB * LL;

    // K0: weight conversion + transpose
    convert_weights_kernel<<<(HH * HH + 255) / 256, 256, 0, stream>>>(gw1, gw2, w1t, w2t);
    // zero the pooled accumulator
    zero_kernel<<<(BB * HH + 255) / 256, 256, 0, stream>>>(h_pool, BB * HH);
    // K1: fused WMMA GNN + mean pool (2048 workgroups x 512 threads)
    gnn_pool_kernel<<<(BB * NN) / 32, 512, 0, stream>>>(X, w1t, gb1, w2t, gb2, h_pool);
    // K2: mu / logvar / z
    head_kernel<<<(BB * LL + 255) / 256, 256, 0, stream>>>(
        h_pool, mu_w, mu_b, lv_w, lv_b, eps, mu_out, lv_out, z);
    // K3: decoder hidden
    dec_z_kernel<<<(BB * HH + 255) / 256, 256, 0, stream>>>(z, z_w, z_b, hd);
    // K4: per-batch node vector + edge scalar
    edge_node_kernel<<<BB, 256, 0, stream>>>(hd, nd_w, nd_b, e1_w, e1_b, e2_w, e2_b,
                                             nodevec, e_scal);
    // K5a: broadcast node logits
    bcast_nodes_kernel<<<((long)BB * NN * TT + 255) / 256, 256, 0, stream>>>(
        nodevec, node_logits);
    // K5b: strict-lower-triangle edge probabilities (float4 stores)
    bcast_edges_kernel<<<((long)BB * NN * (NN / 4) + 255) / 256, 256, 0, stream>>>(
        e_scal, (float4*)edge_probs);
}
